// MateLinAttn_56985626083384
// MI455X (gfx1250) — matrix operand order repaired and checked by reading, not compiled, not run on hardware
//
#include <hip/hip_runtime.h>

// Problem constants (from reference)
#define T_DIM   512
#define B_DIM   4
#define D_DIM   256
#define DFF_DIM 1024

typedef __attribute__((ext_vector_type(16))) __bf16          v16bf;
typedef __attribute__((ext_vector_type(8)))  float           v8f;
typedef __attribute__((ext_vector_type(8)))  unsigned short  us8;
typedef __attribute__((ext_vector_type(4)))  unsigned int    u32x4;
typedef __attribute__((ext_vector_type(8)))  unsigned int    u32x8;

// GEMM tiling: 256 threads = 8 waves; block tile 128x128, K-step 32 (bf16).
// Wave grid 2x4; each wave owns a 64x32 sub-tile = 4x2 WMMA 16x16 tiles.
#define TM 128
#define TN 128
#define TK 32
#define PITCH 40   // LDS row pitch in bf16 elems (32 + 8 pad -> bank spread)

__device__ __forceinline__ unsigned short f2bf(float f) {
  unsigned int u = __float_as_uint(f);
  return (unsigned short)((u + 0x7FFFu + ((u >> 16) & 1u)) >> 16); // RNE
}
__device__ __forceinline__ float gelu_f(float x) {
  float x3 = x * x * x;
  return 0.5f * x * (1.0f + tanhf(0.7978845608028654f * (x + 0.044715f * x3)));
}
__device__ __forceinline__ float elu1_f(float x) {  // elu(x)+1
  return x > 0.0f ? (x + 1.0f) : __expf(x);
}

// ---- Tensor Data Mover: load one 128x32-bf16 tile into LDS with padding ----
// D# group0: count=1, lds_addr, global tile address, type=2 ("image").
// D# group1: data_size=2B; pad_enable, pad_interval=3 (16 DWORDs = one 64B
// row), pad_amount=3 (4 DWORDs = 16B) -> LDS row pitch 80B == PITCH shorts.
// tile_dim0=32 elems, tile_dim1=128 rows, tensor_dim0_stride=ld elems.
// Tensor dims set huge: our tiles never cross the tensor edge (no OOB).
// Issued wave-uniformly by wave 0; tracked with TENSORcnt.
__device__ __forceinline__ void tdm_tile(const unsigned short* gbase, int ld,
                                         int row0, int kt,
                                         unsigned short* lds_tile) {
  const unsigned long long ga =
      (unsigned long long)(uintptr_t)(gbase + (long)row0 * ld + kt);
  const unsigned int lds = (unsigned int)(uintptr_t)lds_tile;
  u32x4 g0;
  g0[0] = 1u;                                                // count=1
  g0[1] = lds;                                               // lds_addr
  g0[2] = (unsigned int)ga;                                  // global lo
  g0[3] = ((unsigned int)(ga >> 32) & 0x01FFFFFFu) | (2u << 30);  // hi | type=2
  u32x8 g1;
  g1[0] = 0x06D10000u;            // data_size=1(2B), pad_en, intv=16dw, amt=4dw
  g1[1] = 0xFFFF0000u;            // tensor_dim0[15:0]=0xFFFF (huge)
  g1[2] = 0xFFFF0FFFu;            // tensor_dim0 hi=0x0FFF | tensor_dim1 lo
  g1[3] = (32u << 16) | 0x0FFFu;  // tensor_dim1 hi | tile_dim0=32
  g1[4] = 128u;                   // tile_dim1=128, tile_dim2=0
  g1[5] = (unsigned int)ld;       // tensor_dim0_stride (elems)
  g1[6] = 0u;
  g1[7] = 0u;
  asm volatile("tensor_load_to_lds %0, %1" :: "s"(g0), "s"(g1) : "memory");
}

// ---- Per-lane async global->LDS: one 128x32-bf16 tile (8 KB) ----
// 512 chunks of 16 B; 256 threads x 2 chunks. Tracked with ASYNCcnt.
__device__ __forceinline__ void async_tile(const unsigned short* gbase, int ld,
                                           int row0, int kt,
                                           unsigned short* lds_tile, int tid) {
#pragma unroll
  for (int i = 0; i < 2; ++i) {
    const int c   = tid + i * 256;     // 0..511
    const int row = c >> 2;            // 0..127
    const int kc  = (c & 3) * 8;       // bf16 elem offset: 0,8,16,24
    const unsigned long long g =
        (unsigned long long)(uintptr_t)(gbase + (long)(row0 + row) * ld + kt + kc);
    const unsigned int l =
        (unsigned int)(uintptr_t)(lds_tile + row * PITCH + kc);
    asm volatile("global_load_async_to_lds_b128 %0, %1, off"
                 :: "v"(l), "v"(g) : "memory");
  }
}

// C[M,N](f32, optional) / Cbf[M,N](bf16, optional) =
//   epilogue( A[m][k](bf16) @ Bt[n][k](bf16)^T + bias ), batched by blockIdx.z.
// mode: 0 bias only, 1 gelu, 2 elu+1, 3 causal mask (col>row -> 0, no bias).
__global__ __launch_bounds__(256) void wmma_gemm_bf16_kernel(
    const unsigned short* __restrict__ A, const unsigned short* __restrict__ Bt,
    const float* __restrict__ bias, float* __restrict__ C,
    unsigned short* __restrict__ Cbf,
    int K, int lda, int ldb, int ldc, int ldcbf,
    long sA, long sB, long sC, long sCbf,
    int mode, int accum)
{
  const int bz = blockIdx.z;
  A  += (long)bz * sA;
  Bt += (long)bz * sB;
  if (C)   C   += (long)bz * sC;
  if (Cbf) Cbf += (long)bz * sCbf;

  const int tid  = threadIdx.x;
  const int lane = tid & 31;
  const int l16  = lane & 15;
  const int hs   = lane >> 4;      // half-wave select (WMMA layouts)
  const int wv   = tid >> 5;
  const int wm   = wv >> 2;        // 0..1
  const int wn   = wv & 3;         // 0..3
  const int M0   = blockIdx.y * TM;
  const int N0   = blockIdx.x * TN;

  // Fully-masked causal tile: zero-fill outputs and exit.
  if (mode == 3 && N0 > M0 + (TM - 1)) {
    for (int i = tid; i < TM * TN; i += 256) {
      const int r = M0 + i / TN, c = N0 + (i % TN);
      if (Cbf) Cbf[(long)r * ldcbf + c] = 0;
      if (C)   C[(long)r * ldc + c] = 0.0f;
    }
    return;
  }

  __shared__ unsigned short Al[2][TM * PITCH];
  __shared__ unsigned short Bl[2][TM * PITCH];

  v8f acc[4][2];
#pragma unroll
  for (int i = 0; i < 4; ++i)
#pragma unroll
    for (int j = 0; j < 2; ++j)
#pragma unroll
      for (int r = 0; r < 8; ++r) acc[i][j][r] = 0.0f;

  // Double-buffered pipeline: A tiles via TDM (wave 0), B tiles via per-lane
  // async copies (all waves).
  if (wv == 0) tdm_tile(A, lda, M0, 0, Al[0]);
  async_tile(Bt, ldb, N0, 0, Bl[0], tid);
  int p = 0;
  for (int kt = 0; kt < K; kt += TK) {
    if (wv == 0) __builtin_amdgcn_s_wait_tensorcnt(0);
    asm volatile("s_wait_asynccnt 0x0" ::: "memory");
    __syncthreads();
    if (kt + TK < K) {
      if (wv == 0) tdm_tile(A, lda, M0, kt + TK, Al[p ^ 1]);
      async_tile(Bt, ldb, N0, kt + TK, Bl[p ^ 1], tid);
    }

    // B fragments (lanes 0-15: K=0..15; lanes 16-31: K=16..31)
    v16bf bfrag[2];
#pragma unroll
    for (int ni = 0; ni < 2; ++ni) {
      const int n  = wn * 32 + ni * 16 + l16;
      const int kb = hs * 16;
      union { v16bf v; us8 h[2]; } u;
      u.h[0] = *(const us8*)&Bl[p][n * PITCH + kb];
      u.h[1] = *(const us8*)&Bl[p][n * PITCH + kb + 8];
      bfrag[ni] = u.v;
    }
    // A fragments (lanes 0-15: K=0..15; lanes 16-31: K=16..31)
#pragma unroll
    for (int mi = 0; mi < 4; ++mi) {
      const int m  = wm * 64 + mi * 16 + l16;
      const int kh = hs * 16;
      union { v16bf v; us8 h[2]; } u;
      u.h[0] = *(const us8*)&Al[p][m * PITCH + kh];
      u.h[1] = *(const us8*)&Al[p][m * PITCH + kh + 8];
      const v16bf afrag = u.v;
#pragma unroll
      for (int ni = 0; ni < 2; ++ni) {
        acc[mi][ni] = __builtin_amdgcn_wmma_f32_16x16x32_bf16(
            false, afrag, false, bfrag[ni], (short)0, acc[mi][ni],
            false, false);
      }
    }
    p ^= 1;
  }

  // Epilogue: C/D layout — lane<16: N=l16, M=r ; lane>=16: N=l16, M=r+8.
#pragma unroll
  for (int mi = 0; mi < 4; ++mi) {
#pragma unroll
    for (int ni = 0; ni < 2; ++ni) {
      const int col = N0 + wn * 32 + ni * 16 + l16;
      const float bval = (bias != nullptr) ? bias[col] : 0.0f;
#pragma unroll
      for (int r = 0; r < 8; ++r) {
        const int row = M0 + wm * 64 + mi * 16 + r + hs * 8;
        float x = acc[mi][ni][r] + bval;
        if (mode == 1)      x = gelu_f(x);
        else if (mode == 2) x = elu1_f(x);
        else if (mode == 3) { if (col > row) x = 0.0f; }
        if (C) {
          const long off = (long)row * ldc + col;
          if (accum) x += C[off];
          C[off] = x;
        }
        if (Cbf) Cbf[(long)row * ldcbf + col] = f2bf(x);
      }
    }
  }
}

// dst[i] = bf16(src[i])
__global__ __launch_bounds__(256) void conv_kernel(
    unsigned short* __restrict__ dst, const float* __restrict__ src, int n) {
  const int i = blockIdx.x * 256 + threadIdx.x;
  if (i < n) dst[i] = f2bf(src[i]);
}

// dst[b][n][k] = bf16(src[b*sSrc + k*ldsrc + n])  (build [n][k] B-operands)
__global__ __launch_bounds__(256) void tconv_kernel(
    unsigned short* __restrict__ dst, const float* __restrict__ src,
    int Kd, int Nd, int ldsrc, long sSrc) {
  const long i = (long)blockIdx.x * 256 + threadIdx.x;
  const int b = blockIdx.y;
  if (i >= (long)Nd * Kd) return;
  const int n = (int)(i / Kd), k = (int)(i % Kd);
  dst[(long)b * Nd * Kd + i] = f2bf(src[(long)b * sSrc + (long)k * ldsrc + n]);
}

// z[t,b,d] = z0[b,d] + cumsum_t k[t,b,d]   (f32, exact recurrence)
__global__ __launch_bounds__(256) void cumsum_kernel(
    float* __restrict__ z, const float* __restrict__ k,
    const float* __restrict__ z0) {
  const int idx = blockIdx.x * 256 + threadIdx.x;   // b*D + d, < B*D
  if (idx >= B_DIM * D_DIM) return;
  float a = z0[idx];
  for (int t = 0; t < T_DIM; ++t) {
    const long off = (long)t * B_DIM * D_DIM + idx;
    a += k[off];
    z[off] = a;
  }
}

// den[row] = dot(z[row,:], q[row,:]) ; one wave32 per row
__global__ __launch_bounds__(256) void den_dot_kernel(
    float* __restrict__ den, const float* __restrict__ z,
    const float* __restrict__ q) {
  const int w = threadIdx.x >> 5, lane = threadIdx.x & 31;
  const int row = blockIdx.x * 8 + w;
  const float* zr = z + (long)row * D_DIM;
  const float* qr = q + (long)row * D_DIM;
  float s = 0.0f;
#pragma unroll
  for (int i = 0; i < D_DIM / 32; ++i) s += zr[lane + i * 32] * qr[lane + i * 32];
#pragma unroll
  for (int o = 16; o > 0; o >>= 1) s += __shfl_xor(s, o, 32);
  if (lane == 0) den[row] = s;
}

// out = (num + init_emb) / max(den, 1e-6)   (in place over d_out)
__global__ __launch_bounds__(256) void finalize_kernel(
    float* __restrict__ out, const float* __restrict__ den,
    const float* __restrict__ init_emb) {
  const int i = blockIdx.x * 256 + threadIdx.x;
  const int row = i / D_DIM, col = i % D_DIM;
  out[i] = (out[i] + init_emb[col]) / fmaxf(den[row], 1e-6f);
}

extern "C" void kernel_launch(void* const* d_in, const int* in_sizes, int n_in,
                              void* d_out, int out_size, void* d_ws, size_t ws_size,
                              hipStream_t stream)
{
  const float* inputs   = (const float*)d_in[0];
  const float* S0       = (const float*)d_in[1];
  const float* z0       = (const float*)d_in[2];
  const float* obs_emb  = (const float*)d_in[3];
  const float* Wfc      = (const float*)d_in[4];
  const float* bfc      = (const float*)d_in[5];
  const float* Wproj    = (const float*)d_in[6];
  const float* bproj    = (const float*)d_in[7];
  const float* Wk       = (const float*)d_in[8];
  const float* bk       = (const float*)d_in[9];
  const float* Wv       = (const float*)d_in[10];
  const float* bv       = (const float*)d_in[11];
  const float* Wq       = (const float*)d_in[12];
  const float* bq       = (const float*)d_in[13];
  const float* init_emb = (const float*)d_in[14];
  float* out = (float*)d_out;

  const int M  = T_DIM * B_DIM;   // 2048 rows
  const int BD = B_DIM * D_DIM;   // 1024

  // ---- carve workspace (all region sizes are 16B multiples) ----
  char* w = (char*)d_ws;
  float* kbuf = (float*)w;  w += (size_t)M * D_DIM * 4;
  float* qbuf = (float*)w;  w += (size_t)M * D_DIM * 4;
  float* vbuf = (float*)w;  w += (size_t)M * D_DIM * 4;
  float* zbuf = (float*)w;  w += (size_t)M * D_DIM * 4;
  float* den  = (float*)w;  w += (size_t)M * 4;
  unsigned short* inbf   = (unsigned short*)w; w += (size_t)M * D_DIM * 2;
  unsigned short* obsbf  = (unsigned short*)w; w += (size_t)M * D_DIM * 2;
  unsigned short* WfcT   = (unsigned short*)w; w += (size_t)2 * D_DIM * DFF_DIM * 2;
  unsigned short* WprojT = (unsigned short*)w; w += (size_t)2 * DFF_DIM * D_DIM * 2;
  unsigned short* WkT    = (unsigned short*)w; w += (size_t)D_DIM * D_DIM * 2;
  unsigned short* WvT    = (unsigned short*)w; w += (size_t)D_DIM * D_DIM * 2;
  unsigned short* WqT    = (unsigned short*)w; w += (size_t)D_DIM * D_DIM * 2;
  unsigned short* S0T    = (unsigned short*)w; w += (size_t)B_DIM * D_DIM * D_DIM * 2;
  unsigned short* hbf    = (unsigned short*)w; w += (size_t)M * DFF_DIM * 2;
  unsigned short* xabf   = (unsigned short*)w; w += (size_t)M * D_DIM * 2;
  unsigned short* xbbf   = (unsigned short*)w; w += (size_t)M * D_DIM * 2;
  unsigned short* kbf    = (unsigned short*)w; w += (size_t)M * D_DIM * 2;
  unsigned short* qbf    = (unsigned short*)w; w += (size_t)M * D_DIM * 2;
  unsigned short* vTbf   = (unsigned short*)w; w += (size_t)B_DIM * D_DIM * T_DIM * 2;
  unsigned short* scbf   = (unsigned short*)w; w += (size_t)B_DIM * T_DIM * T_DIM * 2;

  dim3 blk(256);

  // ---- one-time bf16 conversion / B-operand transposition ----
  conv_kernel<<<dim3(M * D_DIM / 256), blk, 0, stream>>>(inbf, inputs, M * D_DIM);
  conv_kernel<<<dim3(M * D_DIM / 256), blk, 0, stream>>>(obsbf, obs_emb, M * D_DIM);
  tconv_kernel<<<dim3(D_DIM * DFF_DIM / 256, 2), blk, 0, stream>>>(
      WfcT, Wfc, D_DIM, DFF_DIM, DFF_DIM, (long)D_DIM * DFF_DIM);
  tconv_kernel<<<dim3(DFF_DIM * D_DIM / 256, 2), blk, 0, stream>>>(
      WprojT, Wproj, DFF_DIM, D_DIM, D_DIM, (long)DFF_DIM * D_DIM);
  tconv_kernel<<<dim3(D_DIM * D_DIM / 256, 1), blk, 0, stream>>>(
      WkT, Wk, D_DIM, D_DIM, D_DIM, 0);
  tconv_kernel<<<dim3(D_DIM * D_DIM / 256, 1), blk, 0, stream>>>(
      WvT, Wv, D_DIM, D_DIM, D_DIM, 0);
  tconv_kernel<<<dim3(D_DIM * D_DIM / 256, 1), blk, 0, stream>>>(
      WqT, Wq, D_DIM, D_DIM, D_DIM, 0);
  tconv_kernel<<<dim3(D_DIM * D_DIM / 256, B_DIM), blk, 0, stream>>>(
      S0T, S0, D_DIM, D_DIM, D_DIM, (long)D_DIM * D_DIM);

  // ---- embedder: 2 x [ gelu(x Wfc + bfc) Wproj + bproj ] ----
  wmma_gemm_bf16_kernel<<<dim3(DFF_DIM / TN, M / TM, 1), blk, 0, stream>>>(
      inbf, WfcT, bfc, nullptr, hbf,
      D_DIM, D_DIM, D_DIM, 0, DFF_DIM, 0, 0, 0, 0, 1, 0);
  wmma_gemm_bf16_kernel<<<dim3(D_DIM / TN, M / TM, 1), blk, 0, stream>>>(
      hbf, WprojT, bproj, nullptr, xabf,
      DFF_DIM, DFF_DIM, DFF_DIM, 0, D_DIM, 0, 0, 0, 0, 0, 0);
  wmma_gemm_bf16_kernel<<<dim3(DFF_DIM / TN, M / TM, 1), blk, 0, stream>>>(
      xabf, WfcT + (size_t)DFF_DIM * D_DIM, bfc + DFF_DIM, nullptr, hbf,
      D_DIM, D_DIM, D_DIM, 0, DFF_DIM, 0, 0, 0, 0, 1, 0);
  wmma_gemm_bf16_kernel<<<dim3(D_DIM / TN, M / TM, 1), blk, 0, stream>>>(
      hbf, WprojT + (size_t)D_DIM * DFF_DIM, bproj + D_DIM, nullptr, xbbf,
      DFF_DIM, DFF_DIM, DFF_DIM, 0, D_DIM, 0, 0, 0, 0, 0, 0);

  // ---- k = elu(x Wk + bk)+1 ; v = x Wv + bv ; q = elu(obs Wq + bq)+1 ----
  wmma_gemm_bf16_kernel<<<dim3(D_DIM / TN, M / TM, 1), blk, 0, stream>>>(
      xbbf, WkT, bk, kbuf, kbf,
      D_DIM, D_DIM, D_DIM, D_DIM, D_DIM, 0, 0, 0, 0, 2, 0);
  wmma_gemm_bf16_kernel<<<dim3(D_DIM / TN, M / TM, 1), blk, 0, stream>>>(
      xbbf, WvT, bv, vbuf, nullptr,
      D_DIM, D_DIM, D_DIM, D_DIM, 0, 0, 0, 0, 0, 0, 0);
  wmma_gemm_bf16_kernel<<<dim3(D_DIM / TN, M / TM, 1), blk, 0, stream>>>(
      obsbf, WqT, bq, qbuf, qbf,
      D_DIM, D_DIM, D_DIM, D_DIM, D_DIM, 0, 0, 0, 0, 2, 0);

  // ---- vT[b][j][s] = bf16(v[s,b,j]) for num-GEMM B operand ----
  tconv_kernel<<<dim3(D_DIM * T_DIM / 256, B_DIM), blk, 0, stream>>>(
      vTbf, vbuf, T_DIM, D_DIM, BD, (long)D_DIM);

  // ---- scores_b = causal_mask( Q_b K_b^T ), batched over B ----
  wmma_gemm_bf16_kernel<<<dim3(T_DIM / TN, T_DIM / TM, B_DIM), blk, 0, stream>>>(
      qbf, kbf, nullptr, nullptr, scbf,
      D_DIM, BD, BD, 0, T_DIM,
      (long)D_DIM, (long)D_DIM, 0, (long)T_DIM * T_DIM, 3, 0);

  // ---- num = scores_b @ V_b -> d_out (strided (T,B,D)) ----
  wmma_gemm_bf16_kernel<<<dim3(D_DIM / TN, T_DIM / TM, B_DIM), blk, 0, stream>>>(
      scbf, vTbf, nullptr, out, nullptr,
      T_DIM, T_DIM, T_DIM, BD, 0,
      (long)T_DIM * T_DIM, (long)D_DIM * T_DIM, (long)D_DIM, 0, 0, 0);

  // ---- num += Q_b @ S0[b] (fidelity term; S0 is zero at setup) ----
  wmma_gemm_bf16_kernel<<<dim3(D_DIM / TN, T_DIM / TM, B_DIM), blk, 0, stream>>>(
      qbf, S0T, nullptr, out, nullptr,
      D_DIM, BD, D_DIM, BD, 0,
      (long)D_DIM, (long)D_DIM * D_DIM, (long)D_DIM, 0, 0, 1);

  // ---- exact f32 denominator: z = z0 + cumsum(k); den = q . z ----
  cumsum_kernel<<<dim3(BD / 256), blk, 0, stream>>>(zbuf, kbuf, z0);
  den_dot_kernel<<<dim3(M / 8), blk, 0, stream>>>(den, zbuf, qbuf);
  finalize_kernel<<<dim3(M * D_DIM / 256), blk, 0, stream>>>(out, den, init_emb);
}
